// CLAM_65249143161223
// MI455X (gfx1250) — matrix operand order repaired and checked by reading, not compiled, not run on hardware
//
#include <hip/hip_runtime.h>

typedef __attribute__((ext_vector_type(16))) _Float16     v16h;
typedef __attribute__((ext_vector_type(8)))  float        v8f;
typedef __attribute__((ext_vector_type(8)))  unsigned int v8u;

#define N_ROWS 100000
#define BM     32
#define NBLK   (N_ROWS / BM)   // 3125 (exact)
#define KTILES 32              // K = 1024 = 32 * 32

// ---------------------------------------------------------------------------
// Kernel 0: pre-swizzle Wc|W1|W2 (f32, row-major KxN) into f16 fragment-major
// buffer. Tile t = ntile*32 + ktile; element = lane*16 + j.
// B-fragment order used here (contiguous k, same as A; not AMD's table order): n = lane&15,
// k = ktile*32 + (lane>=16 ? 16 : 0) + j   (2 f16 per VGPR, low half first).
// ---------------------------------------------------------------------------
__global__ __launch_bounds__(256) void clam_prep(
    const float* __restrict__ Wc, const float* __restrict__ W1,
    const float* __restrict__ W2, unsigned short* __restrict__ wf)
{
  int gid  = blockIdx.x * 256 + threadIdx.x;   // exactly 2048*512 threads
  int tile = gid >> 9;
  int r    = gid & 511;
  int lane = r >> 4;
  int j    = r & 15;
  int nt   = tile >> 5;
  int kt   = tile & 31;
  int nl   = lane & 15;
  int kb   = (lane & 16) ? 16 : 0;
  int k    = kt * 32 + kb + j;
  const float* src; int NC, n;
  if (nt < 32)      { src = Wc; NC = 512; n =  nt       * 16 + nl; }
  else if (nt < 48) { src = W1; NC = 256; n = (nt - 32) * 16 + nl; }
  else              { src = W2; NC = 256; n = (nt - 48) * 16 + nl; }
  float    v  = src[(size_t)k * NC + n];
  _Float16 hv = (_Float16)v;
  wf[gid] = __builtin_bit_cast(unsigned short, hv);
}

// ---------------------------------------------------------------------------
// Kernel 1: fused per-32-row block.
//   h  = relu(x@Wc+bc)   (cols   0..511)
//   g1 = tanh(x@W1+b1)   (cols 512..767)
//   g2 = sigm(x@W2+b2)   (cols 768..1023)
//   a = g1*g2 ; A = a@W3+b3 -> Aout ; partial[blk] = A_blk^T @ h_blk (2x512)
// 256 threads = 8 waves; wave w owns N-tiles w*8..w*8+7 for both 16-row
// M-tiles. x converted f32->f16 ONCE at staging into a conflict-free
// chunk-fragment LDS layout; epilogue reuses the same 64KB as f32 in two
// 16-row passes.
// ---------------------------------------------------------------------------
__global__ __launch_bounds__(256) void clam_main(
    const float* __restrict__ x, const unsigned short* __restrict__ wf,
    const float* __restrict__ bc, const float* __restrict__ b1,
    const float* __restrict__ b2, const float* __restrict__ W3,
    const float* __restrict__ b3,
    float* __restrict__ Aout, float* __restrict__ part)
{
  __shared__ __align__(16) unsigned char smemraw[65536];
  unsigned short* sx = (unsigned short*)smemraw;  // f16 bits, chunk layout
  float*          sf = (float*)smemraw;           // epilogue float view

  const int t   = threadIdx.x;
  const int bid = blockIdx.x;
  const int r0  = bid * BM;

  // stage + convert x (32 rows x 1024) into sx[((k>>3)*32 + row)*8 + (k&7)]
  #pragma unroll 4
  for (int i = 0; i < 128; ++i) {
    int idx = i * 256 + t;
    int row = idx >> 10;
    int k   = idx & 1023;
    float    v  = x[(size_t)(r0 + row) * 1024 + k];
    _Float16 hv = (_Float16)v;
    sx[((k >> 3) * 32 + row) * 8 + (k & 7)] =
        __builtin_bit_cast(unsigned short, hv);
  }
  __syncthreads();

  const int w   = t >> 5;
  const int l   = t & 31;
  const int ln  = l & 15;
  const int hf  = l >> 4;          // half-wave select
  const int nt0 = w * 8;

  v8f acc[2][8] = {};

  for (int ks = 0; ks < KTILES; ++ks) {
    // A-fragments (16-bit A 16x32): element j -> k=kb16+j,
    // kb16=(lane>=16)*16. Two aligned b128 LDS reads.
    int c0 = 4 * ks + 2 * hf;
    v16h Am[2];
    #pragma unroll
    for (int m = 0; m < 2; ++m) {
      int row = m * 16 + ln;
      const uint4 a0 = *(const uint4*)(sx + (c0 * 32 + row) * 8);
      const uint4 a1 = *(const uint4*)(sx + ((c0 + 1) * 32 + row) * 8);
      v8u u;
      u[0] = a0.x; u[1] = a0.y; u[2] = a0.z; u[3] = a0.w;
      u[4] = a1.x; u[5] = a1.y; u[6] = a1.z; u[7] = a1.w;
      Am[m] = __builtin_bit_cast(v16h, u);
    }
    #pragma unroll
    for (int i = 0; i < 8; ++i) {
      size_t off = ((size_t)((nt0 + i) * KTILES + ks)) * 512 + l * 16;
      v16h B = __builtin_bit_cast(v16h, *(const v8u*)(wf + off));
      acc[0][i] = __builtin_amdgcn_wmma_f32_16x16x32_f16(
          false, Am[0], false, B, (short)0, acc[0][i], false, false);
      acc[1][i] = __builtin_amdgcn_wmma_f32_16x16x32_f16(
          false, Am[1], false, B, (short)0, acc[1][i], false, false);
    }
  }

  // epilogue in two 16-row passes, reusing LDS as f32
  float gp[4] = {0.0f, 0.0f, 0.0f, 0.0f};
  #pragma unroll
  for (int p = 0; p < 2; ++p) {
    __syncthreads();   // pass 0: x reads done; pass 1: prior-pass reads done
    // bias + activation; C/D layout: VGPR v -> M = v + 8*(lane>=16), N=lane&15
    #pragma unroll
    for (int i = 0; i < 8; ++i) {
      int nt  = nt0 + i;
      int col = nt * 16 + ln;
      const float* bias; int base, mode;
      if (nt < 32)      { bias = bc; base = 0;   mode = 0; }
      else if (nt < 48) { bias = b1; base = 512; mode = 1; }
      else              { bias = b2; base = 768; mode = 2; }
      float bv = bias[col - base];
      #pragma unroll
      for (int v = 0; v < 8; ++v) {
        float val = acc[p][i][v] + bv;
        if (mode == 0)      val = fmaxf(val, 0.0f);
        else if (mode == 1) val = tanhf(val);
        else                val = 1.0f / (1.0f + expf(-val));
        sf[(v + 8 * hf) * 1024 + col] = val;
      }
    }
    __syncthreads();
    // a = g1 * g2 (over g1 region)
    #pragma unroll
    for (int i = 0; i < 16; ++i) {
      int idx = i * 256 + t;
      int r = idx >> 8, j = idx & 255;
      sf[r * 1024 + 512 + j] *= sf[r * 1024 + 768 + j];
    }
    __syncthreads();
    // A = a @ W3 + b3 (16x2); stash in unused g2 slots of row 0
    if (t < 32) {
      int r = t >> 1, c = t & 1;
      float s = b3[c];
      for (int j = 0; j < 256; ++j) s += sf[r * 1024 + 512 + j] * W3[j * 2 + c];
      Aout[(size_t)(r0 + p * 16 + r) * 2 + c] = s;
      sf[768 + t] = s;
    }
    __syncthreads();
    // accumulate partial G[c][n] += sum_r A[r][c] * h[r][n]
    #pragma unroll
    for (int i = 0; i < 4; ++i) {
      int idx = i * 256 + t;         // 1024 = 2*512
      int c = idx >> 9, n = idx & 511;
      #pragma unroll
      for (int r = 0; r < 16; ++r)
        gp[i] += sf[768 + r * 2 + c] * sf[r * 1024 + n];
    }
  }
  #pragma unroll
  for (int i = 0; i < 4; ++i)
    part[(size_t)bid * 1024 + i * 256 + t] = gp[i];
}

// ---------------------------------------------------------------------------
// Kernel 2: reduce partials -> slide_agg_rep; @Wbag + bbag; softmax/argmax
// ---------------------------------------------------------------------------
__global__ __launch_bounds__(1024) void clam_slide(
    const float* __restrict__ part, const float* __restrict__ Wbag,
    const float* __restrict__ bbag, float* __restrict__ out)
{
  __shared__ float G[1024];
  __shared__ float sc[2];
  int t = threadIdx.x;
  float s = 0.0f;
  for (int i = 0; i < NBLK; ++i) s += part[(size_t)i * 1024 + t];
  G[t] = s;
  __syncthreads();
  if (t < 2) {
    float v = bbag[0];
    for (int n = 0; n < 512; ++n) v += G[t * 512 + n] * Wbag[n];
    sc[t] = v;
  }
  __syncthreads();
  if (t == 0) {
    float s0 = sc[0], s1 = sc[1];
    out[0] = s0; out[1] = s1;
    out[2] = (s1 > s0) ? 1.0f : 0.0f;           // argmax, first-max tie rule
    float m = fmaxf(s0, s1);
    float e0 = expf(s0 - m), e1 = expf(s1 - m);
    float d = e0 + e1;
    out[3] = e0 / d; out[4] = e1 / d;
  }
}

// ---------------------------------------------------------------------------
// Kernel 3: deterministic top-8 / bottom-8 of A[:, bag_label]
// (ties broken by lowest index, matching jax.lax.top_k)
// ---------------------------------------------------------------------------
__global__ __launch_bounds__(256) void clam_topk(
    const float* __restrict__ A, const int* __restrict__ blp,
    int* __restrict__ idxout)
{
  __shared__ float cv[2048];
  __shared__ int   ci[2048];
  const int t  = threadIdx.x;
  const int bl = blp[0];
  for (int pass = 0; pass < 2; ++pass) {
    float lv[8]; int li[8];
    #pragma unroll
    for (int j = 0; j < 8; ++j) { lv[j] = -3.0e38f; li[j] = 0x7fffffff; }
    for (int i = t; i < N_ROWS; i += 256) {
      float v = A[(size_t)i * 2 + bl];
      if (pass) v = -v;
      if (v > lv[7]) {                 // strict > keeps earlier index on tie
        int j = 7;
        while (j > 0 && v > lv[j - 1]) { lv[j] = lv[j-1]; li[j] = li[j-1]; --j; }
        lv[j] = v; li[j] = i;
      }
    }
    #pragma unroll
    for (int j = 0; j < 8; ++j) { cv[t * 8 + j] = lv[j]; ci[t * 8 + j] = li[j]; }
    __syncthreads();
    if (t == 0) {
      for (int s = 0; s < 8; ++s) {
        float bv = -3.5e38f; int bi = 0x7fffffff, bp = 0;
        for (int p = 0; p < 2048; ++p) {
          float v = cv[p]; int id = ci[p];
          if (v > bv || (v == bv && id < bi)) { bv = v; bi = id; bp = p; }
        }
        cv[bp] = -3.5e38f;
        idxout[pass * 8 + s] = bi;
      }
    }
    __syncthreads();
  }
}

// ---------------------------------------------------------------------------
// Kernel 4: per selected row, recompute h (fp32) and instance softmax logits
// ---------------------------------------------------------------------------
__global__ __launch_bounds__(256) void clam_inst(
    const float* __restrict__ x, const float* __restrict__ Wc,
    const float* __restrict__ bc, const float* __restrict__ Wins,
    const float* __restrict__ bins, const int* __restrict__ idxb,
    float* __restrict__ out)
{
  __shared__ float xr[1024];
  __shared__ float hs[512];
  __shared__ float ls[2];
  const int t = threadIdx.x;
  const int b = blockIdx.x;
  const int row = idxb[b];
  #pragma unroll
  for (int i = 0; i < 4; ++i)
    xr[i * 256 + t] = x[(size_t)row * 1024 + i * 256 + t];
  __syncthreads();
  #pragma unroll
  for (int rep = 0; rep < 2; ++rep) {
    int n = rep * 256 + t;
    float s = bc[n];
    for (int k = 0; k < 1024; ++k) s += xr[k] * Wc[(size_t)k * 512 + n];
    hs[n] = fmaxf(s, 0.0f);
  }
  __syncthreads();
  if (t < 2) {
    float s = bins[t];
    for (int n = 0; n < 512; ++n) s += hs[n] * Wins[n * 2 + t];
    ls[t] = s;
  }
  __syncthreads();
  if (t == 0) {
    float l0 = ls[0], l1 = ls[1];
    float m  = fmaxf(l0, l1);
    float e0 = expf(l0 - m), e1 = expf(l1 - m);
    float d  = e0 + e1;
    out[5 + b * 2 + 0] = e0 / d;
    out[5 + b * 2 + 1] = e1 / d;
    out[37 + b] = (b < 8) ? 1.0f : 0.0f;   // ins_labels
  }
}

// ---------------------------------------------------------------------------
// d_out layout (53 f32): [0:2] slide_score, [2] Y_hat, [3:5] Y_prob,
// [5:37] ins_logits (16x2), [37:53] ins_labels.
// ws layout: wf16 2MB | A 800KB | idx | partials 12.8MB (~16MB total)
// ---------------------------------------------------------------------------
extern "C" void kernel_launch(void* const* d_in, const int* in_sizes, int n_in,
                              void* d_out, int out_size, void* d_ws, size_t ws_size,
                              hipStream_t stream) {
  const float* x    = (const float*)d_in[0];
  const float* Wc   = (const float*)d_in[1];
  const float* bc   = (const float*)d_in[2];
  const float* W1   = (const float*)d_in[3];
  const float* b1   = (const float*)d_in[4];
  const float* W2   = (const float*)d_in[5];
  const float* b2   = (const float*)d_in[6];
  const float* W3   = (const float*)d_in[7];
  const float* b3   = (const float*)d_in[8];
  const float* Wbag = (const float*)d_in[9];
  const float* bbag = (const float*)d_in[10];
  const float* Wins = (const float*)d_in[11];
  const float* bins = (const float*)d_in[12];
  const int*   bl   = (const int*)d_in[13];
  float* out = (float*)d_out;

  char* ws = (char*)d_ws;
  unsigned short* wf   = (unsigned short*)(ws);
  float*          Aarr = (float*)(ws + (2u << 20));
  int*            idxb = (int*)(ws + (2u << 20) + 800000);
  float*          part = (float*)(ws + (3u << 20));

  clam_prep <<<4096, 256, 0, stream>>>(Wc, W1, W2, wf);
  clam_main <<<NBLK, 256, 0, stream>>>(x, wf, bc, b1, b2, W3, b3, Aarr, part);
  clam_slide<<<1, 1024, 0, stream>>>(part, Wbag, bbag, out);
  clam_topk <<<1, 256, 0, stream>>>(Aarr, bl, idxb);
  clam_inst <<<16, 256, 0, stream>>>(x, Wc, bc, Wins, bins, idxb, out);
}
